// GAT_52673478918844
// MI455X (gfx1250) — compile-verified, matrix operand order checked by reading, not run on hardware
//
#include <hip/hip_runtime.h>

// ---------------------------------------------------------------------------
// Types for CDNA5 WMMA (wave32): bf16 A/B fragments, f32 accumulator
// ---------------------------------------------------------------------------
typedef __bf16 bf16;
typedef __attribute__((ext_vector_type(16))) bf16  v16bf;
typedef __attribute__((ext_vector_type(8)))  bf16  v8bf;
typedef __attribute__((ext_vector_type(8)))  float v8f;

static __device__ __forceinline__ unsigned short f2bf(float f) {
  unsigned int u = __float_as_uint(f);
  u += 0x7fffu + ((u >> 16) & 1u);   // round-to-nearest-even
  return (unsigned short)(u >> 16);
}

static __device__ __forceinline__ void atomicMaxF(float* addr, float val) {
  if (val >= 0.0f) {
    atomicMax((int*)addr, __float_as_int(val));
  } else {
    atomicMin((unsigned int*)addr, __float_as_uint(val));
  }
}

// ---------------------------------------------------------------------------
// Utility kernels
// ---------------------------------------------------------------------------
__global__ void k_fill(float* p, long long n, float v) {
  long long i = (long long)blockIdx.x * blockDim.x + threadIdx.x;
  if (i < n) p[i] = v;
}

__global__ void k_cvt_bf16(const float* __restrict__ s, unsigned short* __restrict__ d,
                           long long n) {
  long long i = (long long)blockIdx.x * blockDim.x + threadIdx.x;
  if (i < n) d[i] = f2bf(s[i]);
}

// W: K x C row-major (fp32)  ->  Wt: C x K row-major (bf16)  [i.e. col-major W]
__global__ void k_cvt_transpose_bf16(const float* __restrict__ W,
                                     unsigned short* __restrict__ Wt, int K, int C) {
  int i = blockIdx.x * blockDim.x + threadIdx.x;
  if (i >= K * C) return;
  int k = i / C, c = i % C;
  Wt[c * K + k] = f2bf(W[i]);
}

// ---------------------------------------------------------------------------
// bf16 WMMA GEMM with LDS-staged B:
//   C[M x NC] = A[M x K] * B[K x NC],  Bt given as NC x K bf16 (B col-major),
//   K == 128.  Entire Bt (NC*256 bytes, <= 32KB) is async-copied to LDS once
//   per block (global_load_async_to_lds_b128, ASYNCcnt), then every wave
//   computes a 16-row x 64-col stripe: one A fragment reused across 4 B
//   fragments -> 16 v_wmma_f32_16x16x32_bf16 per wave, B fed from LDS.
// Fragment layout (ISA 7.12.2, 16-bit A 16x32 / B 32x16):
//   lane l: row/col = l&15 ; kbase = (l>>4)*8
//   elements 0..7  -> K = kbase+0..7 ; elements 8..15 -> K = kbase+16..23
// ---------------------------------------------------------------------------
__global__ void k_wmma_gemm_bf16_lds(const bf16* __restrict__ A,
                                     const bf16* __restrict__ Bt,
                                     float* __restrict__ C, int M, int NC) {
  constexpr int K = 128;
  __shared__ __align__(16) bf16 sB[128 * 128];     // 32 KB max (NC=128)

  // ---- Stage Bt into LDS with async copies (16B per lane per issue) ------
  {
    const unsigned totalBytes = (unsigned)NC * K * 2;        // 32768 or 16384
    unsigned ldsOff = (unsigned)(size_t)(&sB[0]) + threadIdx.x * 16u;
    unsigned long long gAddr =
        (unsigned long long)(size_t)((const char*)Bt) + threadIdx.x * 16u;
    for (unsigned off = threadIdx.x * 16u; off < totalBytes;
         off += (unsigned)blockDim.x * 16u) {
      asm volatile("global_load_async_to_lds_b128 %0, %1, off"
                   :: "v"(ldsOff), "v"(gAddr) : "memory");
      ldsOff += (unsigned)blockDim.x * 16u;
      gAddr  += (unsigned)blockDim.x * 16u;
    }
    asm volatile("s_wait_asynccnt 0x0" ::: "memory");
  }
  __syncthreads();

  // ---- Tile assignment: one wave = 16 rows x 64 cols ---------------------
  const int wavesPerBlock = blockDim.x >> 5;
  const int colGroups = NC >> 6;                    // 64-col groups (2 or 1)
  int wave = blockIdx.x * wavesPerBlock + ((int)threadIdx.x >> 5);
  int rowTile  = wave / colGroups;
  int colGroup = wave % colGroups;
  if (rowTile >= (M >> 4)) return;                  // whole-wave exit

  int lane = (int)threadIdx.x & 31;
  int r  = lane & 15;
  int hi = lane >> 4;

  const bf16* arow = A + (long long)(rowTile * 16 + r) * K;
  const bf16* bbase = sB + (colGroup * 64 + r) * K; // + c*16*K per col tile

  v8f acc0, acc1, acc2, acc3;
#pragma unroll
  for (int i = 0; i < 8; ++i) { acc0[i] = 0.f; acc1[i] = 0.f; acc2[i] = 0.f; acc3[i] = 0.f; }

#pragma unroll
  for (int k0 = 0; k0 < K; k0 += 32) {
    int kb = k0 + hi * 8;
    v8bf alo = *(const v8bf*)(arow + kb);
    v8bf ahi = *(const v8bf*)(arow + kb + 16);
    v16bf a;
#pragma unroll
    for (int i = 0; i < 8; ++i) { a[i] = alo[i]; a[i + 8] = ahi[i]; }

#pragma unroll
    for (int c = 0; c < 4; ++c) {
      const bf16* bcol = bbase + c * 16 * K;
      v8bf blo = *(const v8bf*)(bcol + kb);
      v8bf bhi = *(const v8bf*)(bcol + kb + 16);
      v16bf b;
#pragma unroll
      for (int i = 0; i < 8; ++i) { b[i] = blo[i]; b[i + 8] = bhi[i]; }
      v8f& acc = (c == 0) ? acc0 : (c == 1) ? acc1 : (c == 2) ? acc2 : acc3;
      acc = __builtin_amdgcn_wmma_f32_16x16x32_bf16(
          false, a, false, b, (short)0, acc, false, false);
    }
  }

  // C tile: lane l holds column (l&15), rows (l>>4)*8 + v  (v=0..7)
  float* cbase = C + (long long)(rowTile * 16 + hi * 8) * NC + colGroup * 64 + r;
#pragma unroll
  for (int c = 0; c < 4; ++c) {
    const v8f& acc = (c == 0) ? acc0 : (c == 1) ? acc1 : (c == 2) ? acc2 : acc3;
    float* cp = cbase + c * 16;
#pragma unroll
    for (int v = 0; v < 8; ++v) cp[(long long)v * NC] = acc[v];
  }
}

// ---------------------------------------------------------------------------
// Attention coefficients:  el[n,h] = feat[n,h,:].al[h,:]  (same for er)
// ---------------------------------------------------------------------------
__global__ void k_attn_coeff(const float* __restrict__ feat,
                             const float* __restrict__ al, const float* __restrict__ ar,
                             float* __restrict__ el, float* __restrict__ er,
                             long long NH, int H, int D) {
  long long t = (long long)blockIdx.x * blockDim.x + threadIdx.x;
  if (t >= NH) return;
  int h = (int)(t % H);
  const float* f = feat + t * D;
  const float* a = al + h * D;
  const float* b = ar + h * D;
  float sl = 0.0f, sr = 0.0f;
  for (int d = 0; d < D; ++d) { sl += f[d] * a[d]; sr += f[d] * b[d]; }
  el[t] = sl; er[t] = sr;
}

// ---------------------------------------------------------------------------
// Edge pass kernels
// ---------------------------------------------------------------------------
__global__ void k_edge_logit_max(const int* __restrict__ src, const int* __restrict__ dst,
                                 const float* __restrict__ el, const float* __restrict__ er,
                                 float* __restrict__ e, float* __restrict__ m,
                                 long long E, int H) {
  long long t = (long long)blockIdx.x * blockDim.x + threadIdx.x;
  if (t >= E * H) return;
  long long i = t / H;
  int h = (int)(t % H);
  int s = src[i], d = dst[i];
  float x = el[(long long)s * H + h] + er[(long long)d * H + h];
  x = x > 0.0f ? x : 0.2f * x;          // leaky_relu(0.2)
  e[t] = x;
  atomicMaxF(&m[(long long)d * H + h], x);
}

__global__ void k_fix_m(float* __restrict__ m, long long n) {
  long long i = (long long)blockIdx.x * blockDim.x + threadIdx.x;
  if (i >= n) return;
  float v = m[i];
  if (!(fabsf(v) < __builtin_huge_valf())) m[i] = 0.0f;   // isolated nodes
}

__global__ void k_edge_exp_sum(const int* __restrict__ dst, float* __restrict__ e,
                               const float* __restrict__ m, float* __restrict__ den,
                               long long E, int H) {
  long long t = (long long)blockIdx.x * blockDim.x + threadIdx.x;
  if (t >= E * H) return;
  long long i = t / H;
  int h = (int)(t % H);
  int d = dst[i];
  float ex = __expf(e[t] - m[(long long)d * H + h]);
  e[t] = ex;
  atomicAdd(&den[(long long)d * H + h], ex);
}

__global__ void k_edge_alpha(const int* __restrict__ src, const int* __restrict__ dst,
                             float* __restrict__ e, const float* __restrict__ den,
                             const float* __restrict__ feat, long long E, int H, int D) {
  long long t = (long long)blockIdx.x * blockDim.x + threadIdx.x;
  if (t >= E * H) return;
  long long i = t / H;
  int h = (int)(t % H);
  int d = dst[i];
  e[t] = e[t] / fmaxf(den[(long long)d * H + h], 1e-9f);
  // Warm L2 for the upcoming message-scatter gather of feat[src] row.
  __builtin_prefetch(feat + ((long long)src[i] * H + h) * D, 0, 1);
}

// msg scatter: one thread per (edge, head, 4-wide feature quad)
__global__ void k_edge_msg(const int* __restrict__ src, const int* __restrict__ dst,
                           const float* __restrict__ feat, const float* __restrict__ alpha,
                           float* __restrict__ rst, long long E, int H, int D) {
  int DQ = D >> 2;
  long long total = E * (long long)H * DQ;
  long long t = (long long)blockIdx.x * blockDim.x + threadIdx.x;
  if (t >= total) return;
  long long i = t / (H * DQ);
  int rem = (int)(t % (H * DQ));
  int h = rem / DQ, q = rem % DQ;
  int s = src[i], d = dst[i];
  float a = alpha[i * H + h];
  const float4 f = *(const float4*)(feat + ((long long)s * H + h) * D + q * 4);
  float* o = rst + ((long long)d * H + h) * D + q * 4;
  atomicAdd(o + 0, f.x * a);
  atomicAdd(o + 1, f.y * a);
  atomicAdd(o + 2, f.z * a);
  atomicAdd(o + 3, f.w * a);
}

// layer-1 epilogue: h1 = elu(rst + b1), emitted directly as bf16 for GEMM2
__global__ void k_bias_elu_bf16(const float* __restrict__ rst, const float* __restrict__ b,
                                unsigned short* __restrict__ out, long long n, int F) {
  long long i = (long long)blockIdx.x * blockDim.x + threadIdx.x;
  if (i >= n) return;
  float v = rst[i] + b[(int)(i % F)];
  v = v > 0.0f ? v : (__expf(v) - 1.0f);
  out[i] = f2bf(v);
}

// layer-2 epilogue: out += b2 (mean over 1 head is identity)
__global__ void k_bias_out(float* __restrict__ out, const float* __restrict__ b,
                           long long n, int F) {
  long long i = (long long)blockIdx.x * blockDim.x + threadIdx.x;
  if (i >= n) return;
  out[i] += b[(int)(i % F)];
}

// ---------------------------------------------------------------------------
// Launch
// ---------------------------------------------------------------------------
extern "C" void kernel_launch(void* const* d_in, const int* in_sizes, int n_in,
                              void* d_out, int out_size, void* d_ws, size_t ws_size,
                              hipStream_t stream) {
  (void)n_in; (void)out_size; (void)ws_size;
  const int IN = 128, H1 = 8, HID = 16, F1 = H1 * HID /*128*/, OUT = 64;
  const int N = in_sizes[0] / IN;
  const long long E = in_sizes[1];

  const float* x   = (const float*)d_in[0];
  const int*   src = (const int*)  d_in[1];
  const int*   dst = (const int*)  d_in[2];
  const float* W1  = (const float*)d_in[3];
  const float* al1 = (const float*)d_in[4];
  const float* ar1 = (const float*)d_in[5];
  const float* b1  = (const float*)d_in[6];
  const float* W2  = (const float*)d_in[7];
  const float* al2 = (const float*)d_in[8];
  const float* ar2 = (const float*)d_in[9];
  const float* b2  = (const float*)d_in[10];
  float* out = (float*)d_out;

  // Bump allocator over workspace (256B aligned)
  char* w = (char*)d_ws;
  auto alloc = [&](size_t bytes) -> void* {
    void* p = (void*)w;
    w += (bytes + 255) & ~(size_t)255;
    return p;
  };
  unsigned short* Xb    = (unsigned short*)alloc((size_t)N * IN * 2);   // reused as h1b
  unsigned short* W1t   = (unsigned short*)alloc((size_t)IN * F1 * 2);
  unsigned short* W2t   = (unsigned short*)alloc((size_t)F1 * OUT * 2);
  float* feat1 = (float*)alloc((size_t)N * F1 * 4);                     // reused as feat2
  float* el1   = (float*)alloc((size_t)N * H1 * 4);
  float* er1   = (float*)alloc((size_t)N * H1 * 4);
  float* m1    = (float*)alloc((size_t)N * H1 * 4);
  float* den1  = (float*)alloc((size_t)N * H1 * 4);
  float* e1    = (float*)alloc((size_t)E * H1 * 4);                     // reused as e2
  float* rst1  = (float*)alloc((size_t)N * F1 * 4);
  float* el2   = (float*)alloc((size_t)N * 4);
  float* er2   = (float*)alloc((size_t)N * 4);
  float* m2    = (float*)alloc((size_t)N * 4);
  float* den2  = (float*)alloc((size_t)N * 4);

  const int T = 256;
  auto G = [](long long n) { return (unsigned)((n + 255) / 256); };
  const float NEG_INF = -__builtin_huge_valf();

  // ---- Layer 1 -----------------------------------------------------------
  k_cvt_bf16<<<G((long long)N * IN), T, 0, stream>>>(x, Xb, (long long)N * IN);
  k_cvt_transpose_bf16<<<G(IN * F1), T, 0, stream>>>(W1, W1t, IN, F1);
  k_cvt_transpose_bf16<<<G(F1 * OUT), T, 0, stream>>>(W2, W2t, F1, OUT);

  {
    int totalWaves = (N / 16) * (F1 / 64);          // 64-col groups per wave
    k_wmma_gemm_bf16_lds<<<(totalWaves + 7) / 8, 256, 0, stream>>>(
        (const bf16*)Xb, (const bf16*)W1t, feat1, N, F1);
  }

  k_attn_coeff<<<G((long long)N * H1), T, 0, stream>>>(feat1, al1, ar1, el1, er1,
                                                       (long long)N * H1, H1, HID);
  k_fill<<<G((long long)N * H1), T, 0, stream>>>(m1, (long long)N * H1, NEG_INF);
  k_fill<<<G((long long)N * H1), T, 0, stream>>>(den1, (long long)N * H1, 0.0f);
  k_fill<<<G((long long)N * F1), T, 0, stream>>>(rst1, (long long)N * F1, 0.0f);

  k_edge_logit_max<<<G(E * H1), T, 0, stream>>>(src, dst, el1, er1, e1, m1, E, H1);
  k_fix_m<<<G((long long)N * H1), T, 0, stream>>>(m1, (long long)N * H1);
  k_edge_exp_sum<<<G(E * H1), T, 0, stream>>>(dst, e1, m1, den1, E, H1);
  k_edge_alpha<<<G(E * H1), T, 0, stream>>>(src, dst, e1, den1, feat1, E, H1, HID);
  k_edge_msg<<<G(E * H1 * (HID / 4)), T, 0, stream>>>(src, dst, feat1, e1, rst1, E, H1, HID);

  // h1 = elu(rst1 + b1) -> bf16 (into Xb region, now dead)
  k_bias_elu_bf16<<<G((long long)N * F1), T, 0, stream>>>(rst1, b1, Xb, (long long)N * F1, F1);

  // ---- Layer 2 -----------------------------------------------------------
  float* feat2 = feat1;   // reuse (N x 64 fits in N x 128 region)
  float* e2    = e1;      // reuse (E x 1 fits in E x 8 region)
  {
    int totalWaves = (N / 16) * (OUT / 64);
    k_wmma_gemm_bf16_lds<<<(totalWaves + 7) / 8, 256, 0, stream>>>(
        (const bf16*)Xb, (const bf16*)W2t, feat2, N, OUT);
  }

  k_attn_coeff<<<G(N), T, 0, stream>>>(feat2, al2, ar2, el2, er2, N, 1, OUT);
  k_fill<<<G(N), T, 0, stream>>>(m2, N, NEG_INF);
  k_fill<<<G(N), T, 0, stream>>>(den2, N, 0.0f);
  k_fill<<<G((long long)N * OUT), T, 0, stream>>>(out, (long long)N * OUT, 0.0f);

  k_edge_logit_max<<<G(E), T, 0, stream>>>(src, dst, el2, er2, e2, m2, E, 1);
  k_fix_m<<<G(N), T, 0, stream>>>(m2, N);
  k_edge_exp_sum<<<G(E), T, 0, stream>>>(dst, e2, m2, den2, E, 1);
  k_edge_alpha<<<G(E), T, 0, stream>>>(src, dst, e2, den2, feat2, E, 1, OUT);
  k_edge_msg<<<G(E * (OUT / 4)), T, 0, stream>>>(src, dst, feat2, e2, out, E, 1, OUT);

  k_bias_out<<<G((long long)N * OUT), T, 0, stream>>>(out, b2, (long long)N * OUT, OUT);
}
